// TransformerConvNet_5566277616466
// MI455X (gfx1250) — matrix operand order repaired and checked by reading, not compiled, not run on hardware
//
#include <hip/hip_runtime.h>
#include <hip/hip_bf16.h>
#include <math.h>

#define NN    20000     // nodes
#define MPAD  20032     // nodes padded to multiple of 64 (wave M-tile)
#define EE    320000    // edges
#define FIN0  256
#define HIDC  128       // per-head channels
#define NHEAD 4
#define WID   512       // HEADS*HID
#define NCLS  64
#define QW    2048      // Q|K|V|S concatenated output width

typedef __attribute__((ext_vector_type(16))) __bf16 v16bf;
typedef __attribute__((ext_vector_type(8)))  __bf16 v8bf;
typedef __attribute__((ext_vector_type(8)))  float  v8f;

// ---------------- helpers ----------------
__device__ inline __bf16 to_bf16(float f) {
  unsigned u = __float_as_uint(f);
  unsigned r = (u + 0x7FFFu + ((u >> 16) & 1u)) >> 16;
  unsigned short s = (unsigned short)r;
  return __builtin_bit_cast(__bf16, s);
}

__device__ inline void atomicMaxF(float* addr, float val) {
  int cur = __float_as_int(*addr);
  while (__int_as_float(cur) < val) {
    int old = atomicCAS((int*)addr, cur, __float_as_int(val));
    if (old == cur) break;
    cur = old;
  }
}

// ---------------- conversion / setup kernels ----------------
__global__ __launch_bounds__(256) void k_cvt_bf16(const float* __restrict__ in,
                                                  __bf16* __restrict__ out, int n) {
  int i = blockIdx.x * blockDim.x + threadIdx.x;
  if (i < n) out[i] = to_bf16(in[i]);
}

// WT[n][k] (n: 0..2047 output col, k: 0..K-1), from 4 row-major [K][512] fp32 weights
__global__ __launch_bounds__(256) void k_build_wt(const float* __restrict__ W0,
                                                  const float* __restrict__ W1,
                                                  const float* __restrict__ W2,
                                                  const float* __restrict__ W3,
                                                  __bf16* __restrict__ WT, int K) {
  int i = blockIdx.x * blockDim.x + threadIdx.x;
  int total = QW * K;
  if (i >= total) return;
  int n = i / K, kk = i - n * K;
  const float* W = (n < 512) ? W0 : (n < 1024) ? W1 : (n < 1536) ? W2 : W3;
  WT[i] = to_bf16(W[(size_t)kk * WID + (n & 511)]);
}

__global__ __launch_bounds__(256) void k_build_bias(const float* __restrict__ b0,
                                                    const float* __restrict__ b1,
                                                    const float* __restrict__ b2,
                                                    const float* __restrict__ b3,
                                                    float* __restrict__ bcat) {
  int i = blockIdx.x * blockDim.x + threadIdx.x;
  if (i >= QW) return;
  const float* b = (i < 512) ? b0 : (i < 1024) ? b1 : (i < 1536) ? b2 : b3;
  bcat[i] = b[i & 511];
}

// WT[n][k] from row-major W[K][Nc]
__global__ __launch_bounds__(256) void k_build_wt_lin(const float* __restrict__ W,
                                                      __bf16* __restrict__ WT, int K, int Nc) {
  int i = blockIdx.x * blockDim.x + threadIdx.x;
  if (i >= K * Nc) return;
  int n = i / K, kk = i - n * K;
  WT[i] = to_bf16(W[(size_t)kk * Nc + n]);
}

__global__ __launch_bounds__(256) void k_init(float* __restrict__ hagg,
                                              float* __restrict__ amax,
                                              float* __restrict__ denom) {
  int i = blockIdx.x * blockDim.x + threadIdx.x;
  if (i < NN * WID) hagg[i] = 0.f;
  if (i < NN * NHEAD) { amax[i] = -3.0e38f; denom[i] = 0.f; }
}

// ---------------- WMMA GEMM: C[M][Nc] = A[M][K] * BT[Nc][K]^T + bias ----------------
// One wave computes a 64x32 tile of C as 4x2 WMMA subtiles (register blocked):
// per K-step of 32 -> 12 b128 fragment loads, 8 v_wmma_f32_16x16x32_bf16.
// A must have at least roundup(M,64) allocated rows; stores are guarded by row < M.
__global__ __launch_bounds__(256) void k_wmma_gemm(const __bf16* __restrict__ A,
                                                   const __bf16* __restrict__ BT,
                                                   const float* __restrict__ bias,
                                                   float* __restrict__ C,
                                                   int M, int K, int Nc) {
  int wid  = (blockIdx.x * blockDim.x + threadIdx.x) >> 5;
  int lane = threadIdx.x & 31;
  int gn = Nc >> 5;                 // N tiles of 32 per wave
  int gm = (M + 63) >> 6;           // M tiles of 64 per wave
  if (wid >= gm * gn) return;
  int tm = wid / gn, tn = wid - tm * gn;
  int row0 = tm << 6, col0 = tn << 5;
  int m = lane & 15, half = lane >> 4;

  // A fragment (16-bit A 16x32): lane (m,half) holds row m, K = half*16+{0..15}
  const __bf16* a0 = A  + (size_t)(row0 + m) * K + half * 16;
  // B fragment (32x16): lanes 0-15 hold K=0..15 of column m, lanes 16-31 hold K=16..31
  const __bf16* b0 = BT + (size_t)(col0 + m) * K + half * 16;
  size_t step16 = (size_t)16 * K;   // advance 16 rows/cols

  v8f acc[8];
#pragma unroll
  for (int i = 0; i < 8; ++i) acc[i] = (v8f){};

  for (int k0 = 0; k0 < K; k0 += 32) {
    v16bf a[4], b[2];
#pragma unroll
    for (int mi = 0; mi < 4; ++mi) {
      const __bf16* ap = a0 + (size_t)mi * step16 + k0;
      v8bf lo = *(const v8bf*)ap;
      v8bf hi = *(const v8bf*)(ap + 8);
      a[mi] = __builtin_shufflevector(lo, hi, 0,1,2,3,4,5,6,7,8,9,10,11,12,13,14,15);
    }
#pragma unroll
    for (int ni = 0; ni < 2; ++ni)
      b[ni] = *(const v16bf*)(b0 + (size_t)ni * step16 + k0);
#pragma unroll
    for (int mi = 0; mi < 4; ++mi)
#pragma unroll
      for (int ni = 0; ni < 2; ++ni)
        acc[mi * 2 + ni] = __builtin_amdgcn_wmma_f32_16x16x32_bf16(
            false, a[mi], false, b[ni], (short)0, acc[mi * 2 + ni], false, false);
  }

  // C layout per subtile: VGPR r -> M = half*8 + r, N = m
#pragma unroll
  for (int mi = 0; mi < 4; ++mi) {
#pragma unroll
    for (int ni = 0; ni < 2; ++ni) {
      int col = col0 + ni * 16 + m;
      float bb = bias[col];
      int rbase = row0 + mi * 16 + half * 8;
      float* cp = C + (size_t)rbase * Nc + col;
#pragma unroll
      for (int r = 0; r < 8; ++r)
        if (rbase + r < M) cp[(size_t)r * Nc] = acc[mi * 2 + ni][r] + bb;
    }
  }
}

// ---------------- edge kernels ----------------
// one wave per edge: alpha[e][h] = dot(q[dst][h], k[src][h]) / sqrt(HID); amax via atomicMax
__global__ __launch_bounds__(256) void k_edge_logits(const float* __restrict__ qkvs,
                                                     const long long* __restrict__ ei,
                                                     float* __restrict__ alpha,
                                                     float* __restrict__ amax) {
  int wid  = (blockIdx.x * blockDim.x + threadIdx.x) >> 5;
  int lane = threadIdx.x & 31;
  if (wid >= EE) return;
  int src = (int)ei[wid];
  int dst = (int)ei[EE + wid];
  int head = lane >> 3;   // 8 lanes per head
  int sub  = lane & 7;    // 16 channels per lane
  const float* q = qkvs + (size_t)dst * QW +        head * HIDC + sub * 16;
  const float* k = qkvs + (size_t)src * QW + 512 +  head * HIDC + sub * 16;
  float s = 0.f;
#pragma unroll
  for (int i = 0; i < 16; ++i) s += q[i] * k[i];
  s += __shfl_xor(s, 1, 32);
  s += __shfl_xor(s, 2, 32);
  s += __shfl_xor(s, 4, 32);
  s *= 0.08838834764831845f;   // 1/sqrt(128)
  if (sub == 0) {
    alpha[(size_t)wid * NHEAD + head] = s;
    atomicMaxF(&amax[dst * NHEAD + head], s);
  }
}

// exp(alpha - amax[dst]) in place + denom accumulation
__global__ __launch_bounds__(256) void k_edge_exp(const long long* __restrict__ ei,
                                                  float* __restrict__ alpha,
                                                  const float* __restrict__ amax,
                                                  float* __restrict__ denom) {
  int i = blockIdx.x * blockDim.x + threadIdx.x;
  if (i >= EE * NHEAD) return;
  int e = i >> 2, h = i & 3;
  int dst = (int)ei[EE + e];
  float v = __expf(alpha[i] - amax[dst * NHEAD + h]);
  alpha[i] = v;
  atomicAdd(&denom[dst * NHEAD + h], v);
}

// one wave per edge: hagg[dst] += (ex/denom[dst]) * v[src]
__global__ __launch_bounds__(256) void k_aggregate(const float* __restrict__ qkvs,
                                                   const long long* __restrict__ ei,
                                                   const float* __restrict__ alpha,
                                                   const float* __restrict__ denom,
                                                   float* __restrict__ hagg) {
  int wid  = (blockIdx.x * blockDim.x + threadIdx.x) >> 5;
  int lane = threadIdx.x & 31;
  if (wid >= EE) return;
  int src = (int)ei[wid];
  int dst = (int)ei[EE + wid];
  int head = lane >> 3;
  float a = alpha[(size_t)wid * NHEAD + head] / denom[dst * NHEAD + head];
  const float* v = qkvs + (size_t)src * QW + 1024 + lane * 16;
  float*       o = hagg + (size_t)dst * WID + lane * 16;
#pragma unroll
  for (int i = 0; i < 16; ++i) atomicAdd(&o[i], a * v[i]);
}

// h = skip(S part of qkvs) + aggregated messages; optional ELU; convert to bf16 for next GEMM
__global__ __launch_bounds__(256) void k_finalize(const float* __restrict__ qkvs,
                                                  const float* __restrict__ hagg,
                                                  __bf16* __restrict__ xb, int do_elu) {
  int i = blockIdx.x * blockDim.x + threadIdx.x;
  if (i >= NN * WID) return;
  int n = i >> 9, c = i & 511;
  float val = qkvs[(size_t)n * QW + 1536 + c] + hagg[i];
  if (do_elu) val = (val > 0.f) ? val : (__expf(val) - 1.f);
  xb[i] = to_bf16(val);
}

// ---------------- host ----------------
static inline int nblk(long long n, int t) { return (int)((n + t - 1) / t); }

extern "C" void kernel_launch(void* const* d_in, const int* in_sizes, int n_in,
                              void* d_out, int out_size, void* d_ws, size_t ws_size,
                              hipStream_t stream) {
  const float* x       = (const float*)d_in[0];
  const long long* ei  = (const long long*)d_in[1];
  const float* Wlin    = (const float*)d_in[26];
  const float* blin    = (const float*)d_in[27];

  // workspace carve (all sizes 256B aligned)
  char* p = (char*)d_ws;
  float*  qkvs  = (float*)p;            p += (size_t)NN * QW * 4;      // 164 MB
  float*  hagg  = (float*)p;            p += (size_t)NN * WID * 4;     // 41 MB
  __bf16* xb    = (__bf16*)p;           p += (size_t)MPAD * WID * 2;   // 20.5 MB (padded rows)
  float*  alpha = (float*)p;            p += (size_t)EE * NHEAD * 4;   // 5.1 MB
  float*  amax  = (float*)p;            p += (size_t)NN * NHEAD * 4;
  float*  denom = (float*)p;            p += (size_t)NN * NHEAD * 4;
  __bf16* WT    = (__bf16*)p;           p += (size_t)QW * WID * 2;     // 2 MB
  float*  bcat  = (float*)p;            p += (size_t)QW * 4;
  __bf16* WTlin = (__bf16*)p;           p += (size_t)NCLS * WID * 2;

  const int T = 256;

  // layer-0 input -> bf16 (row stride = 256)
  k_cvt_bf16<<<nblk((long long)NN * FIN0, T), T, 0, stream>>>(x, xb, NN * FIN0);

  for (int l = 0; l < 3; ++l) {
    int K = (l == 0) ? FIN0 : WID;
    const float* Wq = (const float*)d_in[2 + 8 * l + 0];
    const float* bq = (const float*)d_in[2 + 8 * l + 1];
    const float* Wk = (const float*)d_in[2 + 8 * l + 2];
    const float* bk = (const float*)d_in[2 + 8 * l + 3];
    const float* Wv = (const float*)d_in[2 + 8 * l + 4];
    const float* bv = (const float*)d_in[2 + 8 * l + 5];
    const float* Ws = (const float*)d_in[2 + 8 * l + 6];
    const float* bs = (const float*)d_in[2 + 8 * l + 7];

    k_build_wt<<<nblk((long long)QW * K, T), T, 0, stream>>>(Wq, Wk, Wv, Ws, WT, K);
    k_build_bias<<<nblk(QW, T), T, 0, stream>>>(bq, bk, bv, bs, bcat);
    k_init<<<nblk((long long)NN * WID, T), T, 0, stream>>>(hagg, amax, denom);

    // fused Q|K|V|S projection: [N,K] x [K,2048], 64x32 per wave
    long long waves = (long long)((NN + 63) / 64) * (QW / 32);
    k_wmma_gemm<<<nblk(waves * 32, T), T, 0, stream>>>(xb, WT, bcat, qkvs, NN, K, QW);

    k_edge_logits<<<nblk((long long)EE * 32, T), T, 0, stream>>>(qkvs, ei, alpha, amax);
    k_edge_exp<<<nblk((long long)EE * NHEAD, T), T, 0, stream>>>(ei, alpha, amax, denom);
    k_aggregate<<<nblk((long long)EE * 32, T), T, 0, stream>>>(qkvs, ei, alpha, denom, hagg);
    k_finalize<<<nblk((long long)NN * WID, T), T, 0, stream>>>(qkvs, hagg, xb, l < 2 ? 1 : 0);
  }

  // final linear: [N,512] x [512,64] -> d_out (fp32), stores guarded by row < NN
  k_build_wt_lin<<<nblk((long long)WID * NCLS, T), T, 0, stream>>>(Wlin, WTlin, WID, NCLS);
  long long fwaves = (long long)((NN + 63) / 64) * (NCLS / 32);
  k_wmma_gemm<<<nblk(fwaves * 32, T), T, 0, stream>>>(xb, WTlin, blin, (float*)d_out,
                                                      NN, WID, NCLS);
}
